// SoftCodebook_77988016161041
// MI455X (gfx1250) — matrix operand order repaired and checked by reading, not compiled, not run on hardware
//
#include <hip/hip_runtime.h>
#include <hip/hip_bf16.h>

typedef __attribute__((ext_vector_type(16))) _Float16 v16h;
typedef __attribute__((ext_vector_type(8)))  float    v8f;

#define D_DIM 1024
#define K_DIM 1024
#define ROWS_PER_WG 128
#define INV_TAU 1.0f

// ---------------- fragment loaders (CDNA5 16-bit WMMA layouts) ----------------
// A-matrix 16x32 f16: lane m(0..15) rows; lanes<16 hold K {0..7,16..23},
// lanes>=16 hold K {8..15,24..31}. Two 16-byte contiguous segments.
__device__ __forceinline__ v16h load_a_frag(const _Float16* rowPtr, int k0, int half) {
    union { v16h v; uint4 q[2]; } u;
    const _Float16* p = rowPtr + k0 + half * 8;
    u.q[0] = *(const uint4*)(p);
    u.q[1] = *(const uint4*)(p + 16);
    return u.v;
}
// B-matrix 32x16 f16: lane holds one column; lanes<16 hold K {0..7,16..23},
// lanes>=16 hold K {8..15,24..31}. Two 16-byte contiguous segments.
__device__ __forceinline__ v16h load_b_frag(const _Float16* colPtr, int k0, int half) {
    union { v16h v; uint4 q[2]; } u;
    const _Float16* p = colPtr + k0 + half * 8;
    u.q[0] = *(const uint4*)(p);
    u.q[1] = *(const uint4*)(p + 16);
    return u.v;
}

// ---------------- prep: codebook fp32 -> f16 (row-major + transposed), c2[k] ----
__global__ __launch_bounds__(256) void prep_codebook(const float* __restrict__ cb,
                                                     _Float16* __restrict__ cb16,
                                                     _Float16* __restrict__ cbt16,
                                                     float* __restrict__ c2) {
    int k = blockIdx.x;
    int tid = threadIdx.x;
    float partial = 0.0f;
    for (int d = tid; d < D_DIM; d += 256) {
        float v = cb[(long)k * D_DIM + d];
        partial += v * v;
        _Float16 h = (_Float16)v;
        cb16[(long)k * D_DIM + d] = h;          // [k][d]  (B for GEMM1)
        cbt16[(long)d * K_DIM + k] = h;         // [d][k]  (B for GEMM2)
    }
    __shared__ float red[256];
    red[tid] = partial;
    __syncthreads();
    for (int off = 128; off > 0; off >>= 1) {
        if (tid < off) red[tid] += red[tid + off];
        __syncthreads();
    }
    if (tid == 0) c2[k] = red[0];
}

// ---------------- fused: GEMM1 -> online softmax -> GEMM2 ----------------------
__global__ __launch_bounds__(256) void fused_vq(const float* __restrict__ emb,
                                                const float* __restrict__ noise,
                                                const _Float16* __restrict__ cb16,
                                                const _Float16* __restrict__ cbt16,
                                                const float* __restrict__ c2,
                                                float* __restrict__ q_out,
                                                float* __restrict__ logits_out) {
    extern __shared__ char smem[];
    _Float16* sTile = (_Float16*)smem;                        // 128x1024 f16: emb, then weights
    float* sX2 = (float*)(smem + ROWS_PER_WG * D_DIM * 2);    // [128]
    float* sM  = sX2 + ROWS_PER_WG;                           // [128]
    float* sS  = sM  + ROWS_PER_WG;                           // [128]

    const int tid  = threadIdx.x;
    const int wave = tid >> 5;
    const int lane = tid & 31;
    const int half = lane >> 4;
    const int lr   = lane & 15;
    const long rowBase = (long)blockIdx.x * ROWS_PER_WG;

    // ---- Phase A: stage 128x1024 embeddings into LDS as f16, compute x2 ----
    for (int r = 0; r < 16; ++r) {
        int lrow = wave * 16 + r;
        const float* src = emb + (rowBase + lrow) * D_DIM;
        _Float16* dst = sTile + (long)lrow * D_DIM;
        float acc = 0.0f;
#pragma unroll
        for (int j = 0; j < 32; ++j) {
            int d = lane + j * 32;
            float v = src[d];
            acc += v * v;
            dst[d] = (_Float16)v;
        }
        acc += __shfl_xor(acc, 16, 32);
        acc += __shfl_xor(acc, 8, 32);
        acc += __shfl_xor(acc, 4, 32);
        acc += __shfl_xor(acc, 2, 32);
        acc += __shfl_xor(acc, 1, 32);
        if (lane == 0) sX2[lrow] = acc;
    }
    __syncthreads();

    // ---- Phase B: GEMM1 (xc) + epilogue (logits, gumbel, online softmax) ----
    float mrun[8], srun[8];
#pragma unroll
    for (int r = 0; r < 8; ++r) { mrun[r] = -3.0e38f; srun[r] = 0.0f; }

    const _Float16* aBase = sTile + (long)(wave * 16 + lr) * D_DIM;

    for (int nb = 0; nb < 16; ++nb) {
        int n0 = nb * 64;
        v8f acc[4] = {};
        const _Float16* bp0 = cb16 + (long)(n0 + 0 * 16 + lr) * D_DIM;
        const _Float16* bp1 = cb16 + (long)(n0 + 1 * 16 + lr) * D_DIM;
        const _Float16* bp2 = cb16 + (long)(n0 + 2 * 16 + lr) * D_DIM;
        const _Float16* bp3 = cb16 + (long)(n0 + 3 * 16 + lr) * D_DIM;
#pragma unroll 4
        for (int dc = 0; dc < 32; ++dc) {
            int d0 = dc * 32;
            v16h a = load_a_frag(aBase, d0, half);
            v16h b0 = load_b_frag(bp0, d0, half);
            acc[0] = __builtin_amdgcn_wmma_f32_16x16x32_f16(false, a, false, b0, (short)0, acc[0], false, false);
            v16h b1 = load_b_frag(bp1, d0, half);
            acc[1] = __builtin_amdgcn_wmma_f32_16x16x32_f16(false, a, false, b1, (short)0, acc[1], false, false);
            v16h b2 = load_b_frag(bp2, d0, half);
            acc[2] = __builtin_amdgcn_wmma_f32_16x16x32_f16(false, a, false, b2, (short)0, acc[2], false, false);
            v16h b3 = load_b_frag(bp3, d0, half);
            acc[3] = __builtin_amdgcn_wmma_f32_16x16x32_f16(false, a, false, b3, (short)0, acc[3], false, false);
        }
        // epilogue: C layout -> row M = r + 8*half, col N = lr
#pragma unroll
        for (int t = 0; t < 4; ++t) {
            int code = n0 + t * 16 + lr;
            float c2v = c2[code];
#pragma unroll
            for (int r = 0; r < 8; ++r) {
                int M = r + half * 8;
                long grow = rowBase + wave * 16 + M;
                float xc = acc[t][r];
                float d2 = sX2[wave * 16 + M] + c2v - 2.0f * xc;
                d2 = fmaxf(d2, 0.0f);
                float logit = -__builtin_sqrtf(d2);
                long gi = grow * K_DIM + code;
                logits_out[gi] = logit;
                float u = noise[gi];
                float g = -__logf(-__logf(u));
                float z = (logit + g) * INV_TAU;
                // row-wise max over the 16 lanes of this half
                float tmax = z;
                tmax = fmaxf(tmax, __shfl_xor(tmax, 1, 32));
                tmax = fmaxf(tmax, __shfl_xor(tmax, 2, 32));
                tmax = fmaxf(tmax, __shfl_xor(tmax, 4, 32));
                tmax = fmaxf(tmax, __shfl_xor(tmax, 8, 32));
                float mnew = fmaxf(mrun[r], tmax);
                float p = __expf(z - mnew);
                p += __shfl_xor(p, 1, 32);
                p += __shfl_xor(p, 2, 32);
                p += __shfl_xor(p, 4, 32);
                p += __shfl_xor(p, 8, 32);
                srun[r] = srun[r] * __expf(mrun[r] - mnew) + p;
                mrun[r] = mnew;
            }
        }
    }
    if (lr == 0) {
#pragma unroll
        for (int r = 0; r < 8; ++r) {
            int M = r + half * 8;
            sM[wave * 16 + M] = mrun[r];
            sS[wave * 16 + M] = srun[r];
        }
    }
    __syncthreads();

    // ---- Phase C: materialize softmax weights as f16 into the SAME LDS tile ----
    for (int r = 0; r < 16; ++r) {
        int lrow = wave * 16 + r;
        long grow = rowBase + lrow;
        float mr = sM[lrow];
        float rs = 1.0f / sS[lrow];
        const float* lg = logits_out + grow * K_DIM;
        const float* nz = noise + grow * K_DIM;
        _Float16* dst = sTile + (long)lrow * K_DIM;
#pragma unroll 4
        for (int j = 0; j < 32; ++j) {
            int c = lane + j * 32;
            float g = -__logf(-__logf(nz[c]));
            float z = (lg[c] + g) * INV_TAU;
            float w = __expf(z - mr) * rs;
            dst[c] = (_Float16)w;
        }
    }
    __syncthreads();

    // ---- Phase D: GEMM2 (quantized = W * codebook), A from LDS, B = cbt16 ----
    const _Float16* aW = sTile + (long)(wave * 16 + lr) * K_DIM;
    for (int db = 0; db < 16; ++db) {
        int d0 = db * 64;
        v8f acc[4] = {};
        const _Float16* bp0 = cbt16 + (long)(d0 + 0 * 16 + lr) * K_DIM;
        const _Float16* bp1 = cbt16 + (long)(d0 + 1 * 16 + lr) * K_DIM;
        const _Float16* bp2 = cbt16 + (long)(d0 + 2 * 16 + lr) * K_DIM;
        const _Float16* bp3 = cbt16 + (long)(d0 + 3 * 16 + lr) * K_DIM;
#pragma unroll 4
        for (int kc = 0; kc < 32; ++kc) {
            int k0 = kc * 32;
            v16h a = load_a_frag(aW, k0, half);
            v16h b0 = load_b_frag(bp0, k0, half);
            acc[0] = __builtin_amdgcn_wmma_f32_16x16x32_f16(false, a, false, b0, (short)0, acc[0], false, false);
            v16h b1 = load_b_frag(bp1, k0, half);
            acc[1] = __builtin_amdgcn_wmma_f32_16x16x32_f16(false, a, false, b1, (short)0, acc[1], false, false);
            v16h b2 = load_b_frag(bp2, k0, half);
            acc[2] = __builtin_amdgcn_wmma_f32_16x16x32_f16(false, a, false, b2, (short)0, acc[2], false, false);
            v16h b3 = load_b_frag(bp3, k0, half);
            acc[3] = __builtin_amdgcn_wmma_f32_16x16x32_f16(false, a, false, b3, (short)0, acc[3], false, false);
        }
#pragma unroll
        for (int t = 0; t < 4; ++t) {
            int dcol = d0 + t * 16 + lr;
#pragma unroll
            for (int r = 0; r < 8; ++r) {
                int M = r + half * 8;
                long grow = rowBase + wave * 16 + M;
                q_out[grow * D_DIM + dcol] = acc[t][r];
            }
        }
    }
}

extern "C" void kernel_launch(void* const* d_in, const int* in_sizes, int n_in,
                              void* d_out, int out_size, void* d_ws, size_t ws_size,
                              hipStream_t stream) {
    const float* emb   = (const float*)d_in[0];   // [4,8192,1024]
    const float* cb    = (const float*)d_in[1];   // [1024,1024]
    const float* noise = (const float*)d_in[2];   // [4,8192,1024]

    const long n_rows = 4L * 8192L;               // 32768
    float* q_out      = (float*)d_out;            // quantized first
    float* logits_out = q_out + n_rows * D_DIM;   // then logits

    // workspace: cb f16 row-major | cb f16 transposed | c2
    _Float16* cb16  = (_Float16*)d_ws;
    _Float16* cbt16 = cb16 + (long)K_DIM * D_DIM;
    float*    c2    = (float*)(cbt16 + (long)K_DIM * D_DIM);

    prep_codebook<<<K_DIM, 256, 0, stream>>>(cb, cb16, cbt16, c2);

    size_t smem = (size_t)ROWS_PER_WG * D_DIM * 2 + 3 * ROWS_PER_WG * sizeof(float);
    int n_wg = (int)(n_rows / ROWS_PER_WG);       // 256
    fused_vq<<<n_wg, 256, smem, stream>>>(emb, noise, cb16, cbt16, c2,
                                          q_out, logits_out);
}
